// ProposalModule_2886218023354
// MI455X (gfx1250) — matrix operand order repaired and checked by reading, not compiled, not run on hardware
//
#include <hip/hip_runtime.h>
#include <hip/hip_bf16.h>

// ---------------------------------------------------------------------------
// ProposalModule (PointNet++ SA + head) for MI455X / gfx1250
//   - FPS: register-resident points/distances, LDS broadcast, wave32 argmax
//   - ball query staged in LDS; features transposed once (f32->f16, coalesced)
//   - grouped MLP + FC layers as f16 WMMA GEMMs (v_wmma_f32_16x16x32_f16)
//   - MT=4 M-tiles per block: each B fragment feeds 4 WMMAs (4 indep accs)
//   - K templated (288/128): full unroll, cross-kt software pipelining
//   - A-tile staging via GLOBAL_LOAD_ASYNC_TO_LDS_B128 (ASYNCcnt path)
//   - max-over-group fused in layer-2 epilogue, head transform fused in c3
// ---------------------------------------------------------------------------

typedef _Float16 v16h __attribute__((ext_vector_type(16)));
typedef _Float16 h8   __attribute__((ext_vector_type(8)));
typedef float    v8f  __attribute__((ext_vector_type(8)));

#define NPTS   8192
#define NPOINT 1024
#define NB     8
#define NSAMP  16
#define RAD2   0.09f
#define INV_R  (1.0f/0.3f)
#define MT     4          // M-tiles (16 rows each) per block
#define USE_ASYNC_LDS 1

// ---------------------------------------------------------------------------
// Pre-swizzle weights into WMMA B-fragment order (f16).
// out index = ((t*KT + kt)*32 + lane)*16 + i ;  B[k][n] = w[n][k_src]
//   n = t*16 + (lane&15) ; k = kt*32 + ((lane>>4)&1)*16 + i
// remap!=0 (layer0 only): h0 column k holds feature c=k+3 for k<256,
//   xyz dim (k-256) for 256<=k<259, zero-pad above  ->  remap k accordingly.
// ---------------------------------------------------------------------------
__global__ void prep_w(const float* __restrict__ w, _Float16* __restrict__ out,
                       int O, int C, int Kpad, int total, int remap) {
  int e = blockIdx.x * blockDim.x + threadIdx.x;
  if (e >= total) return;
  int i    = e & 15;
  int l    = (e >> 4) & 31;
  int rest = e >> 9;
  int ktiles = Kpad >> 5;
  int kt = rest % ktiles;
  int t  = rest / ktiles;
  int n = t * 16 + (l & 15);
  int k = kt * 32 + ((l >> 4) & 1) * 16 + i;
  float v = 0.f;
  if (n < O) {
    if (remap) {
      if (k < 256)      v = w[n * C + (k + 3)];
      else if (k < 259) v = w[n * C + (k - 256)];
    } else if (k < C) {
      v = w[n * C + k];
    }
  }
  out[e] = (_Float16)v;
}

// ---------------------------------------------------------------------------
// Farthest point sampling: one 1024-thread block (32 waves) per batch.
// Each thread owns 8 points: coords + running min-dist in REGISTERS;
// LDS holds xyz only for broadcasting the current farthest point.
// ---------------------------------------------------------------------------
__global__ void fps_kernel(const float* __restrict__ xyz,
                           int* __restrict__ inds, float* __restrict__ new_xyz) {
  extern __shared__ float smem[];
  float* sx = smem;
  float* sy = smem + NPTS;
  float* sz = smem + 2 * NPTS;
  float* rv = smem + 3 * NPTS;            // 32 per-wave best values
  int*   ri = (int*)(smem + 3 * NPTS + 32);
  int* sfar = (int*)(smem + 3 * NPTS + 64);

  const int b = blockIdx.x, tid = threadIdx.x;
  float px[8], py[8], pz[8], pd[8];
#pragma unroll
  for (int p = 0; p < 8; ++p) {
    int j = tid + p * 1024;
    const float* q = xyz + ((size_t)b * NPTS + j) * 3;
    px[p] = q[0]; py[p] = q[1]; pz[p] = q[2]; pd[p] = 1e10f;
    sx[j] = px[p]; sy[j] = py[p]; sz[j] = pz[p];
  }
  __syncthreads();

  const int lane = tid & 31, wv = tid >> 5;
  int far = 0;
  for (int it = 0; it < NPOINT; ++it) {
    if (tid == 0) {
      inds[b * NPOINT + it] = far;
      float* nz = new_xyz + ((size_t)b * NPOINT + it) * 3;
      nz[0] = sx[far]; nz[1] = sy[far]; nz[2] = sz[far];
    }
    const float fx = sx[far], fy = sy[far], fz = sz[far];   // LDS broadcast
    float bv = -1.f; int bi = 0;
#pragma unroll
    for (int p = 0; p < 8; ++p) {
      float dx = px[p] - fx, dy = py[p] - fy, dz = pz[p] - fz;
      float d  = dx * dx + dy * dy + dz * dz;
      float nd = fminf(pd[p], d);
      pd[p] = nd;
      if (nd > bv) { bv = nd; bi = tid + p * 1024; }  // ascending j keeps first
    }
    // wave32 argmax (prefer smaller index on ties, matching jnp.argmax)
#pragma unroll
    for (int off = 16; off; off >>= 1) {
      float ov = __shfl_xor(bv, off, 32);
      int   oi = __shfl_xor(bi, off, 32);
      if (ov > bv || (ov == bv && oi < bi)) { bv = ov; bi = oi; }
    }
    if (lane == 0) { rv[wv] = bv; ri[wv] = bi; }
    __syncthreads();
    if (wv == 0) {
      float bv2 = rv[lane]; int bi2 = ri[lane];
#pragma unroll
      for (int off = 16; off; off >>= 1) {
        float ov = __shfl_xor(bv2, off, 32);
        int   oi = __shfl_xor(bi2, off, 32);
        if (ov > bv2 || (ov == bv2 && oi < bi2)) { bv2 = ov; bi2 = oi; }
      }
      if (lane == 0) *sfar = bi2;
    }
    __syncthreads();
    far = *sfar;
  }
}

// ---------------------------------------------------------------------------
// Ball query: one wave per query (first 16 in-radius indices, ascending),
// wave32 ballot + popcount prefix; xyz staged in LDS per block.
// ---------------------------------------------------------------------------
__global__ void ballquery_kernel(const float* __restrict__ xyz,
                                 const float* __restrict__ new_xyz,
                                 int* __restrict__ idxout) {
  extern __shared__ float smem[];
  float* sx = smem;
  float* sy = smem + NPTS;
  float* sz = smem + 2 * NPTS;
  const int q0  = blockIdx.x * 8;            // 8 waves -> 8 queries, same batch
  const int b   = q0 >> 10;
  const int tid = threadIdx.x;
  for (int j = tid; j < NPTS; j += 256) {
    const float* p = xyz + ((size_t)b * NPTS + j) * 3;
    sx[j] = p[0]; sy[j] = p[1]; sz[j] = p[2];
  }
  __syncthreads();
  const int wv = tid >> 5, lane = tid & 31;
  const int q  = q0 + wv;
  const float* nz = new_xyz + (size_t)q * 3;
  const float qx = nz[0], qy = nz[1], qz = nz[2];
  int* op = idxout + (size_t)q * NSAMP;
  int cnt = 0, first = 0;
  for (int base = 0; base < NPTS && cnt < NSAMP; base += 32) {
    int j = base + lane;
    float dx = sx[j] - qx, dy = sy[j] - qy, dz = sz[j] - qz;
    bool in = (dx * dx + dy * dy + dz * dz) < RAD2;
    unsigned mask = (unsigned)__ballot(in);
    if (cnt == 0 && mask) first = base + __builtin_ctz(mask);
    if (in) {
      int pos = cnt + __builtin_popcount(mask & ((1u << lane) - 1u));
      if (pos < NSAMP) op[pos] = j;
    }
    cnt += __builtin_popcount(mask);
  }
  if (lane < NSAMP && lane >= cnt) op[lane] = (cnt == 0) ? 0 : first;
}

// ---------------------------------------------------------------------------
// Transpose features (B,256,NPTS) f32 -> featT (B,NPTS,256) f16.
// 32x32 LDS tile; reads and writes both coalesced.
// ---------------------------------------------------------------------------
__global__ void transpose_feat(const float* __restrict__ f, _Float16* __restrict__ ft) {
  __shared__ float tile[32][33];
  const int b  = blockIdx.z;
  const int c0 = blockIdx.y * 32;
  const int j0 = blockIdx.x * 32;
  const int tx = threadIdx.x & 31;
  const int ty = threadIdx.x >> 5;     // 0..7
#pragma unroll
  for (int i = 0; i < 4; ++i) {
    int c = c0 + ty + i * 8;
    tile[ty + i * 8][tx] = f[((size_t)b * 256 + c) * NPTS + j0 + tx];
  }
  __syncthreads();
#pragma unroll
  for (int i = 0; i < 4; ++i) {
    int j = j0 + ty + i * 8;
    ft[((size_t)b * NPTS + j) * 256 + c0 + tx] = (_Float16)tile[tx][ty + i * 8];
  }
}

// ---------------------------------------------------------------------------
// Build grouped matrix h0 (f16): M = B*NPOINT*NSAMP rows, 288 cols.
// Column order: [0..255]=features, [256..258]=(xyz-center)/R, [259..287]=0.
// One 16-byte chunk per thread (36 chunks/row) -> fully coalesced copies.
// ---------------------------------------------------------------------------
__global__ void build_h0(const float* __restrict__ xyz,
                         const _Float16* __restrict__ featT,
                         const float* __restrict__ new_xyz,
                         const int* __restrict__ idx,
                         _Float16* __restrict__ h0, int total) {
  int e = blockIdx.x * 256 + threadIdx.x;
  if (e >= total) return;
  int c = e % 36;                  // 16B chunk within row
  int m = e / 36;
  int n = m & 15;
  int g = m >> 4;                  // flat (b*NPOINT + s)
  int b = g >> 10;
  int j = idx[g * NSAMP + n];
  h8 out = {};
  if (c < 32) {
    out = *(const h8*)(featT + ((size_t)b * NPTS + j) * 256 + c * 8);
  } else if (c == 32) {
    const float* p  = xyz + ((size_t)b * NPTS + j) * 3;
    const float* nz = new_xyz + (size_t)g * 3;
    out[0] = (_Float16)((p[0] - nz[0]) * INV_R);
    out[1] = (_Float16)((p[1] - nz[1]) * INV_R);
    out[2] = (_Float16)((p[2] - nz[2]) * INV_R);
  }
  *(h8*)(h0 + (size_t)m * 288 + c * 8) = out;
}

// ---------------------------------------------------------------------------
// WMMA GEMM: D[M x 16*NT] = A[M x K] * Wsw (+epilogue).  K is compile-time.
// Block = 256 thr = 8 waves; wave w owns N-tile w for MT consecutive M-tiles.
// Per k-step: 1 B-fragment load + MT A fragments + MT chained WMMAs.
// mode 0: BN+ReLU -> f16                         (MLP layers 0/1)
// mode 1: BN+ReLU + max over 16-row group -> f16 (MLP layer 2 -> feat)
// mode 2: +bias, BN+ReLU -> f16                  (FC c1/c2)
// mode 3: +bias, head transform -> f32 d_out     (c3, 97 valid channels)
// ---------------------------------------------------------------------------
template <int K>
__global__ void gemm16(const _Float16* __restrict__ A,
                       const _Float16* __restrict__ Wsw,
                       const float* __restrict__ gamma, const float* __restrict__ beta,
                       const float* __restrict__ mean,  const float* __restrict__ var,
                       const float* __restrict__ bias,  const float* __restrict__ new_xyz,
                       const float* __restrict__ msa,
                       _Float16* __restrict__ outh, float* __restrict__ outf,
                       int NT, int mode) {
  constexpr int KT = K >> 5;
  __shared__ _Float16 As[16 * MT * K];
  const int tid   = threadIdx.x;
  const int tile0 = blockIdx.x * MT;         // first 16-row M-tile

  // ---- stage 16*MT contiguous rows of A into LDS ----
#if USE_ASYNC_LDS
  {
    constexpr int nchunk = (16 * MT * K) >> 3;   // 16-byte chunks
    const unsigned ldsbase = (unsigned)(size_t)(&As[0]);
    const _Float16* src = A + (size_t)tile0 * 16 * K;
#pragma unroll 2
    for (int c = tid; c < nchunk; c += 256) {
      unsigned loff = ldsbase + c * 16;
      const void* g = (const void*)(src + c * 8);
      asm volatile("global_load_async_to_lds_b128 %0, %1, off"
                   :: "v"(loff), "v"(g) : "memory");
    }
#if __has_builtin(__builtin_amdgcn_s_wait_asynccnt)
    __builtin_amdgcn_s_wait_asynccnt(0);
#else
    asm volatile("s_wait_asynccnt 0" ::: "memory");
#endif
  }
#else
  {
    const unsigned long long* src =
        (const unsigned long long*)(A + (size_t)tile0 * 16 * K);
    unsigned long long* dst = (unsigned long long*)As;
    constexpr int cnt = (16 * MT * K) >> 2;
    for (int i = tid; i < cnt; i += 256) dst[i] = src[i];
  }
#endif
  __syncthreads();

  const int wv = tid >> 5, lane = tid & 31;
  if (wv >= NT) return;
  const int half = (lane >> 4) & 1;
  const int nn   = lane & 15;

  v8f acc[MT];
#pragma unroll
  for (int t = 0; t < MT; ++t) acc[t] = (v8f){0.f,0.f,0.f,0.f,0.f,0.f,0.f,0.f};

  const _Float16* wbase = Wsw + (((size_t)wv * KT) * 32 + lane) * 16;

#pragma unroll
  for (int kt = 0; kt < KT; ++kt) {
    // B fragment: pre-swizzled, 32B contiguous per lane (coalesced, L2-hot)
    v16h bfrag = *(const v16h*)(wbase + (size_t)kt * 32 * 16);
    const int koff = kt * 32 + half * 16;
    // preload MT A fragments (in-order ds returns -> partial dscnt waits)
    h8 lo[MT], hi[MT];
#pragma unroll
    for (int t = 0; t < MT; ++t) {
      const _Float16* ap = &As[(t * 16 + nn) * K + koff];
      lo[t] = *(const h8*)ap;            // ds_load_b128
      hi[t] = *(const h8*)(ap + 8);      // ds_load_b128
    }
#pragma unroll
    for (int t = 0; t < MT; ++t) {
      v16h a;
#pragma unroll
      for (int i = 0; i < 8; ++i) { a[i] = lo[t][i]; a[i + 8] = hi[t][i]; }
      acc[t] = __builtin_amdgcn_wmma_f32_16x16x32_f16(false, a, false, bfrag,
                                                      (short)0, acc[t], false, false);
    }
  }

  const int o = wv * 16 + nn;
  if (mode == 0 || mode == 2) {
    const float s  = gamma[o] * rsqrtf(var[o] + 1e-5f);
    const float bb = beta[o] - mean[o] * s;
    const float bo = (mode == 2) ? bias[o] : 0.f;
#pragma unroll
    for (int t = 0; t < MT; ++t) {
#pragma unroll
      for (int r = 0; r < 8; ++r) {
        float v = (acc[t][r] + bo) * s + bb;
        v = fmaxf(v, 0.f);
        size_t m = (size_t)(tile0 + t) * 16 + r + 8 * half;
        outh[m * 128 + o] = (_Float16)v;
      }
    }
  } else if (mode == 1) {
    const float s  = gamma[o] * rsqrtf(var[o] + 1e-5f);
    const float bb = beta[o] - mean[o] * s;
#pragma unroll
    for (int t = 0; t < MT; ++t) {
      float vmax = -1e30f;
#pragma unroll
      for (int r = 0; r < 8; ++r) {
        float v = fmaxf(acc[t][r] * s + bb, 0.f);
        vmax = fmaxf(vmax, v);
      }
      vmax = fmaxf(vmax, __shfl_xor(vmax, 16, 32));   // combine row halves
      if (half == 0) outh[(size_t)(tile0 + t) * 128 + o] = (_Float16)vmax;
    }
  } else { // mode 3: head
    if (o < 97) {
#pragma unroll
      for (int t = 0; t < MT; ++t) {
#pragma unroll
        for (int r = 0; r < 8; ++r) {
          size_t m = (size_t)(tile0 + t) * 16 + r + 8 * half; // flat (b*NPOINT+s)
          float v = acc[t][r] + bias[o];
          if (o >= 2 && o < 5)        v += new_xyz[m * 3 + (o - 2)];
          else if (o == 6)            v *= 3.14159265358979323846f; // pi/NH, NH=1
          else if (o >= 25 && o < 79) v *= msa[o - 25];
          outf[m * 97 + o] = v;
        }
      }
    }
  }
}

// ---------------------------------------------------------------------------
extern "C" void kernel_launch(void* const* d_in, const int* in_sizes, int n_in,
                              void* d_out, int out_size, void* d_ws, size_t ws_size,
                              hipStream_t stream) {
  const float* xyz      = (const float*)d_in[0];
  const float* features = (const float*)d_in[1];
  const float* w0 = (const float*)d_in[2];
  const float* g0 = (const float*)d_in[3];
  const float* b0 = (const float*)d_in[4];
  const float* m0 = (const float*)d_in[5];
  const float* v0 = (const float*)d_in[6];
  const float* w1 = (const float*)d_in[7];
  const float* g1 = (const float*)d_in[8];
  const float* b1 = (const float*)d_in[9];
  const float* m1 = (const float*)d_in[10];
  const float* v1 = (const float*)d_in[11];
  const float* w2 = (const float*)d_in[12];
  const float* g2 = (const float*)d_in[13];
  const float* b2 = (const float*)d_in[14];
  const float* m2 = (const float*)d_in[15];
  const float* v2 = (const float*)d_in[16];
  const float* c1w = (const float*)d_in[17];
  const float* c1b = (const float*)d_in[18];
  const float* c2w = (const float*)d_in[19];
  const float* c2b = (const float*)d_in[20];
  const float* c3w = (const float*)d_in[21];
  const float* c3b = (const float*)d_in[22];
  const float* bn1g = (const float*)d_in[23];
  const float* bn1b = (const float*)d_in[24];
  const float* bn1m = (const float*)d_in[25];
  const float* bn1v = (const float*)d_in[26];
  const float* bn2g = (const float*)d_in[27];
  const float* bn2b = (const float*)d_in[28];
  const float* bn2m = (const float*)d_in[29];
  const float* bn2v = (const float*)d_in[30];
  const float* msa  = (const float*)d_in[31];

  char* ws = (char*)d_ws;
  int*       fps_inds = (int*)(ws + 0);                       // 32 KB
  float*     new_xyz  = (float*)(ws + 32768);                 // 96 KB
  int*       idxb     = (int*)(ws + 131072);                  // 512 KB
  _Float16*  Wsw0     = (_Float16*)(ws + 655360);             // 72 KB
  _Float16*  Wsw1     = (_Float16*)(ws + 729088);             // 32 KB
  _Float16*  Wsw2     = (_Float16*)(ws + 761856);
  _Float16*  WswC1    = (_Float16*)(ws + 794624);
  _Float16*  WswC2    = (_Float16*)(ws + 827392);
  _Float16*  WswC3    = (_Float16*)(ws + 860160);             // 28 KB
  _Float16*  h0       = (_Float16*)(ws + (1ull << 20));       // 131072 x 288 f16
  _Float16*  bufA     = (_Float16*)(ws + (1ull << 20) + 75497472ull); // 131072 x 128 f16
  _Float16*  featT    = bufA;  // (B,NPTS,256) f16 = 32MB, dead before layer-0 GEMM
  _Float16*  feat     = (_Float16*)(ws + (1ull << 20) + 75497472ull + 33554432ull);
  _Float16*  net1     = (_Float16*)((char*)feat + 2097152);
  _Float16*  net2     = (_Float16*)((char*)net1 + 2097152);
  float*     out      = (float*)d_out;

  // weight pre-swizzle into WMMA B-fragment order (layer0 with column remap)
  prep_w<<<(36864 + 255) / 256, 256, 0, stream>>>(w0, Wsw0, 128, 259, 288, 36864, 1);
  prep_w<<<64, 256, 0, stream>>>(w1,  Wsw1,  128, 128, 128, 16384, 0);
  prep_w<<<64, 256, 0, stream>>>(w2,  Wsw2,  128, 128, 128, 16384, 0);
  prep_w<<<64, 256, 0, stream>>>(c1w, WswC1, 128, 128, 128, 16384, 0);
  prep_w<<<64, 256, 0, stream>>>(c2w, WswC2, 128, 128, 128, 16384, 0);
  prep_w<<<56, 256, 0, stream>>>(c3w, WswC3,  97, 128, 128, 14336, 0);

  // FPS: 8 blocks x 1024 threads; 96KB LDS (xyz broadcast) + reduce scratch
  fps_kernel<<<NB, 1024, (3 * NPTS + 80) * sizeof(float), stream>>>(xyz, fps_inds, new_xyz);
  // ball query: 8192 queries / 8 per block; 96KB LDS
  ballquery_kernel<<<1024, 256, 3 * NPTS * sizeof(float), stream>>>(xyz, new_xyz, idxb);
  // feature transpose to point-major f16 (coalesced gather source)
  transpose_feat<<<dim3(NPTS / 32, 8, NB), 256, 0, stream>>>(features, featT);
  // grouped matrix h0: 16B chunks, fully coalesced
  const int TOTC = 131072 * 36;
  build_h0<<<(TOTC + 255) / 256, 256, 0, stream>>>(xyz, featT, new_xyz, idxb, h0, TOTC);

  // grouped MLP: 3x WMMA GEMM (M=131072 -> 8192 tiles -> 2048 blocks)
  gemm16<288><<<2048, 256, 0, stream>>>(h0,   Wsw0, g0, b0, m0, v0, nullptr, nullptr, nullptr,
                                        bufA, nullptr, 8, 0);
  gemm16<128><<<2048, 256, 0, stream>>>(bufA, Wsw1, g1, b1, m1, v1, nullptr, nullptr, nullptr,
                                        h0,   nullptr, 8, 0);   // reuse h0 buffer
  gemm16<128><<<2048, 256, 0, stream>>>(h0,   Wsw2, g2, b2, m2, v2, nullptr, nullptr, nullptr,
                                        feat, nullptr, 8, 1);   // fused max-over-group

  // FC head: M=8192 -> 512 tiles -> 128 blocks
  gemm16<128><<<128, 256, 0, stream>>>(feat, WswC1, bn1g, bn1b, bn1m, bn1v, c1b, nullptr, nullptr,
                                       net1, nullptr, 8, 2);
  gemm16<128><<<128, 256, 0, stream>>>(net1, WswC2, bn2g, bn2b, bn2m, bn2v, c2b, nullptr, nullptr,
                                       net2, nullptr, 8, 2);
  gemm16<128><<<128, 256, 0, stream>>>(net2, WswC3, nullptr, nullptr, nullptr, nullptr, c3b,
                                       new_xyz, msa, nullptr, out, 7, 3);

  (void)in_sizes; (void)n_in; (void)out_size; (void)ws_size;
}
